// RetNet_70720931496205
// MI455X (gfx1250) — matrix operand order repaired and checked by reading, not compiled, not run on hardware
//
#include <hip/hip_runtime.h>
#include <hip/hip_bf16.h>
#include <math.h>

typedef __attribute__((ext_vector_type(16))) _Float16 v16h;
typedef __attribute__((ext_vector_type(8)))  _Float16 v8h;
typedef __attribute__((ext_vector_type(8)))  float    v8f;

#define B_     1024
#define EPS_   1e-5f

__device__ __forceinline__ float gelu_exact(float x) {
    return 0.5f * x * (1.0f + erff(x * 0.70710678118654752f));
}

union HV { v16h v; v8h h[2]; };

// A fragment: two contiguous 8-half runs (K = hi*8+0..7 and 16+hi*8+0..7)
__device__ __forceinline__ v16h ldA(const _Float16* arow, int k0) {
    HV a;
    a.h[0] = *(const v8h*)(arow + k0);
    a.h[1] = *(const v8h*)(arow + k0 + 16);
    return a.v;
}
// B fragment from transposed (NxK) weights: two contiguous 8-half runs (K = hi*8+0..7 and 16+hi*8+0..7)
__device__ __forceinline__ v16h ldB(const _Float16* bcol, int k0) {
    HV b;
    b.h[0] = *(const v8h*)(bcol + k0);
    b.h[1] = *(const v8h*)(bcol + k0 + 16);
    return b.v;
}
#define WMMA(ACC, A, Bv) __builtin_amdgcn_wmma_f32_16x16x32_f16(false, (A), false, (Bv), (short)0, (ACC), false, false)

// ---------------------------------------------------------------- conv3d + bias
// X:(B,1,30,13,13) w:(8,1,3,3,3) -> f16 (B,11,11,224), c = ch*28 + d
__global__ void conv3_kernel(const float* __restrict__ X, const float* __restrict__ w,
                             const float* __restrict__ bias, _Float16* __restrict__ out) {
    int idx = blockIdx.x * blockDim.x + threadIdx.x;
    const int total = B_ * 8 * 28 * 11 * 11;
    if (idx >= total) return;
    int x = idx % 11; int t = idx / 11;
    int y = t % 11;  t /= 11;
    int d = t % 28;  t /= 28;
    int ch = t % 8;  int b = t / 8;
    float acc = bias[ch];
#pragma unroll
    for (int cd = 0; cd < 3; ++cd)
#pragma unroll
        for (int cy = 0; cy < 3; ++cy)
#pragma unroll
            for (int cx = 0; cx < 3; ++cx)
                acc += X[(b * 30 + d + cd) * 169 + (y + cy) * 13 + (x + cx)] *
                       w[ch * 27 + cd * 9 + cy * 3 + cx];
    out[(b * 121 + y * 11 + x) * 224 + ch * 28 + d] = (_Float16)acc;
}

// ---------------------------------------------------------------- BN3 stats (8 channels)
__global__ void bn3_stats(const _Float16* __restrict__ a, float* __restrict__ stats) {
    __shared__ float ss[256], sq[256];
    int ch = blockIdx.x;
    const int per = B_ * 121 * 28;
    float s = 0.f, q = 0.f;
    for (int e = threadIdx.x; e < per; e += blockDim.x) {
        int d = e % 28; int r = e / 28;
        float v = (float)a[r * 224 + ch * 28 + d];
        s += v; q += v * v;
    }
    ss[threadIdx.x] = s; sq[threadIdx.x] = q; __syncthreads();
    for (int o = 128; o > 0; o >>= 1) {
        if (threadIdx.x < o) { ss[threadIdx.x] += ss[threadIdx.x + o]; sq[threadIdx.x] += sq[threadIdx.x + o]; }
        __syncthreads();
    }
    if (threadIdx.x == 0) {
        float m = ss[0] / per;
        stats[ch] = m; stats[8 + ch] = sq[0] / per - m * m;
    }
}

__global__ void bn3_apply(_Float16* a, const float* __restrict__ stats,
                          const float* __restrict__ g, const float* __restrict__ b) {
    int idx = blockIdx.x * blockDim.x + threadIdx.x;
    if (idx >= B_ * 121 * 224) return;
    int c = idx % 224; int ch = c / 28;
    float v = ((float)a[idx] - stats[ch]) * rsqrtf(stats[8 + ch] + EPS_) * g[ch] + b[ch];
    a[idx] = (_Float16)fmaxf(v, 0.f);
}

// ---------------------------------------------------------------- conv2 weights -> transposed padded (64 x 2304) f16
__global__ void pack_w2t(const float* __restrict__ w, _Float16* __restrict__ o) {
    int idx = blockIdx.x * blockDim.x + threadIdx.x;
    if (idx >= 64 * 2304) return;
    int k = idx % 2304; int oc = idx / 2304;
    int c = k & 255; int r = k >> 8; int kx = r % 3; int ky = r / 3;
    float v = (c < 224) ? w[((oc * 224 + c) * 3 + ky) * 3 + kx] : 0.f;
    o[oc * 2304 + k] = (_Float16)v;
}

// ---------------------------------------------------------------- conv2 implicit-GEMM WMMA, NT=4
// A fragment double-buffered; B fragments reloaded in place right after use.
__global__ void __launch_bounds__(256, 1)
conv2_wmma(const _Float16* __restrict__ A2, const _Float16* __restrict__ Wt,
           const float* __restrict__ bias, float* __restrict__ C) {
    const int KP = 2304;
    int wave = threadIdx.x >> 5, lane = threadIdx.x & 31;
    long long tile = (long long)blockIdx.x * 8 + wave;
    if (tile >= 82944 / 16) return;              // wave-uniform exit
    int mi = (int)tile;
    int hi = lane >> 4, l15 = lane & 15;
    int row = mi * 16 + l15;
    int b = row / 81, p = row % 81, oy = p / 9, ox = p % 9;
    const _Float16* bc[4];
    bc[0] = Wt + (size_t)l15 * KP + hi * 8;
    bc[1] = bc[0] + 16 * KP;
    bc[2] = bc[0] + 32 * KP;
    bc[3] = bc[0] + 48 * KP;
    int pixBase = b * 121 + oy * 11 + ox;

    auto aPtr = [&](int k0) -> const _Float16* {
        int r = k0 >> 8; int cb = k0 & 255;
        int kx = r % 3, ky = r / 3;
        return A2 + (size_t)(pixBase + ky * 11 + kx) * 224 + cb + hi * 8;
    };

    v8f acc[4];
#pragma unroll
    for (int j = 0; j < 4; ++j) acc[j] = (v8f){};
    v16h a = ldA(aPtr(0), 0);
    v16h bfrag[4];
#pragma unroll
    for (int j = 0; j < 4; ++j) bfrag[j] = ldB(bc[j], 0);

    for (int k0 = 32; k0 < KP; k0 += 32) {
        v16h an = ldA(aPtr(k0), 0);              // next A before this step's WMMAs
#pragma unroll
        for (int j = 0; j < 4; ++j) {
            acc[j] = WMMA(acc[j], a, bfrag[j]);
            bfrag[j] = ldB(bc[j], k0);           // reload in place after use
        }
        a = an;
    }
#pragma unroll
    for (int j = 0; j < 4; ++j) acc[j] = WMMA(acc[j], a, bfrag[j]);

#pragma unroll
    for (int r = 0; r < 8; ++r) {
        int m = mi * 16 + hi * 8 + r;
#pragma unroll
        for (int j = 0; j < 4; ++j)
            C[m * 64 + j * 16 + l15] = acc[j][r] + bias[j * 16 + l15];
    }
}

// ---------------------------------------------------------------- BN2 stats (64 channels)
__global__ void bn2_stats(const float* __restrict__ C, float* __restrict__ stats) {
    __shared__ float ss[256], sq[256];
    int c = blockIdx.x;
    const int per = B_ * 81;
    float s = 0.f, q = 0.f;
    for (int r = threadIdx.x; r < per; r += blockDim.x) {
        float v = C[r * 64 + c]; s += v; q += v * v;
    }
    ss[threadIdx.x] = s; sq[threadIdx.x] = q; __syncthreads();
    for (int o = 128; o > 0; o >>= 1) {
        if (threadIdx.x < o) { ss[threadIdx.x] += ss[threadIdx.x + o]; sq[threadIdx.x] += sq[threadIdx.x + o]; }
        __syncthreads();
    }
    if (threadIdx.x == 0) {
        float m = ss[0] / per;
        stats[64 + c] = m; stats[128 + c] = sq[0] / per - m * m;
    }
}

__global__ void bn2_apply8(const float* __restrict__ C, const float* __restrict__ stats,
                           const float* __restrict__ g, const float* __restrict__ b,
                           float* __restrict__ x) {
    int i = blockIdx.x * blockDim.x + threadIdx.x;
    if (i >= 82944 * 64 / 8) return;
    int base = i * 8; int c0 = base & 63;
    float4 v0 = *(const float4*)(C + base);
    float4 v1 = *(const float4*)(C + base + 4);
    float o[8] = {v0.x, v0.y, v0.z, v0.w, v1.x, v1.y, v1.z, v1.w};
#pragma unroll
    for (int j = 0; j < 8; ++j) {
        int c = c0 + j;
        o[j] = fmaxf((o[j] - stats[64 + c]) * rsqrtf(stats[128 + c] + EPS_) * g[c] + b[c], 0.f);
    }
    *(float4*)(x + base) = make_float4(o[0], o[1], o[2], o[3]);
    *(float4*)(x + base + 4) = make_float4(o[4], o[5], o[6], o[7]);
}

// ---------------------------------------------------------------- LayerNorm (width 64) -> f16
__global__ void ln_kernel(const float* __restrict__ xin, const float* __restrict__ g,
                          const float* __restrict__ b, _Float16* __restrict__ out, int rows) {
    int r = blockIdx.x * blockDim.x + threadIdx.x;
    if (r >= rows) return;
    const float4* p4 = (const float4*)(xin + r * 64);
    float buf[64];
    float s = 0.f;
#pragma unroll
    for (int i = 0; i < 16; ++i) {
        float4 v = p4[i];
        buf[4*i] = v.x; buf[4*i+1] = v.y; buf[4*i+2] = v.z; buf[4*i+3] = v.w;
        s += v.x + v.y + v.z + v.w;
    }
    float mu = s * (1.f / 64.f);
    float q = 0.f;
#pragma unroll
    for (int i = 0; i < 64; ++i) { float d = buf[i] - mu; q += d * d; }
    float rstd = rsqrtf(q * (1.f / 64.f) + EPS_);
#pragma unroll
    for (int i = 0; i < 8; ++i) {
        v8h o;
#pragma unroll
        for (int j = 0; j < 8; ++j)
            o[j] = (_Float16)((buf[8*i+j] - mu) * rstd * g[8*i+j] + b[8*i+j]);
        *(v8h*)(out + r * 64 + 8 * i) = o;
    }
}

// ---------------------------------------------------------------- pack [Wq|Wk|Wv|Wg]^T -> (256 x 64) f16
__global__ void pack_wcat_t(const float* __restrict__ wq, const float* __restrict__ wk,
                            const float* __restrict__ wv, const float* __restrict__ wg,
                            int l, _Float16* __restrict__ o) {
    int idx = blockIdx.x * blockDim.x + threadIdx.x;
    if (idx >= 64 * 256) return;
    int col = idx % 256; int d = idx / 256;
    float v;
    if (col < 192) {
        const float* w = (col < 64) ? wq : ((col < 128) ? wk : wv);
        int cc = col & 63; int h = cc >> 4; int j = cc & 15;
        v = w[((l * 4 + h) * 64 + d) * 16 + j];
    } else {
        v = wg[(l * 64 + d) * 64 + (col - 192)];
    }
    o[col * 64 + d] = (_Float16)v;
}

// transpose + cast: in (K x N) f32 -> out (N x K) f16
__global__ void tr_cast(const float* __restrict__ in, _Float16* __restrict__ out, int K, int N) {
    int idx = blockIdx.x * blockDim.x + threadIdx.x;
    if (idx >= K * N) return;
    int n = idx % N; int k = idx / N;
    out[n * K + k] = (_Float16)in[idx];
}

// 8-wide f32 -> f16 cast
__global__ void cast8(const float* __restrict__ in, _Float16* __restrict__ out, int n8) {
    int i = blockIdx.x * blockDim.x + threadIdx.x;
    if (i >= n8) return;
    int base = i * 8;
    float4 v0 = *(const float4*)(in + base);
    float4 v1 = *(const float4*)(in + base + 4);
    v8h o;
    o[0]=(_Float16)v0.x; o[1]=(_Float16)v0.y; o[2]=(_Float16)v0.z; o[3]=(_Float16)v0.w;
    o[4]=(_Float16)v1.x; o[5]=(_Float16)v1.y; o[6]=(_Float16)v1.z; o[7]=(_Float16)v1.w;
    *(v8h*)(out + base) = o;
}

// ---------------------------------------------------------------- generic WMMA GEMM, Bt is (N x K) f16
// NT n-tiles per wave; A double-buffered, B reloaded in place.
// Requires M%16==0, N%(16*NT)==0, K%32==0.
template <int NT>
__global__ void __launch_bounds__(256, 1)
gemm_wmma(const _Float16* __restrict__ A, const _Float16* __restrict__ Bt,
          const float* __restrict__ bias, const float* __restrict__ resid,
          float* __restrict__ Cf, _Float16* __restrict__ Ch,
          int M, int N, int K, int mode) {
    int wave = threadIdx.x >> 5, lane = threadIdx.x & 31;
    int groupsN = (N >> 4) / NT;
    long long total = (long long)(M >> 4) * groupsN;
    long long t = (long long)blockIdx.x * 8 + wave;
    if (t >= total) return;                      // wave-uniform exit, EXEC all-ones
    int mi = (int)(t / groupsN), ng = (int)(t % groupsN);
    int hi = lane >> 4, l15 = lane & 15;
    int row = mi * 16 + l15;
    const _Float16* arow = A + (size_t)row * K + hi * 8;
    const _Float16* bcol[NT];
#pragma unroll
    for (int j = 0; j < NT; ++j)
        bcol[j] = Bt + (size_t)((ng * NT + j) * 16 + l15) * K + hi * 8;

    v8f acc[NT];
#pragma unroll
    for (int j = 0; j < NT; ++j) acc[j] = (v8f){};
    v16h a = ldA(arow, 0);
    v16h b[NT];
#pragma unroll
    for (int j = 0; j < NT; ++j) b[j] = ldB(bcol[j], 0);

    for (int k0 = 32; k0 < K; k0 += 32) {
        __builtin_prefetch((const void*)(arow + k0 + 128), 0, 0);
        v16h an = ldA(arow, k0);                 // next A before this step's WMMAs
#pragma unroll
        for (int j = 0; j < NT; ++j) {
            acc[j] = WMMA(acc[j], a, b[j]);
            b[j] = ldB(bcol[j], k0);             // reload in place after use
        }
        a = an;
    }
#pragma unroll
    for (int j = 0; j < NT; ++j) acc[j] = WMMA(acc[j], a, b[j]);

#pragma unroll
    for (int j = 0; j < NT; ++j) {
        int col = (ng * NT + j) * 16 + l15;
#pragma unroll
        for (int r = 0; r < 8; ++r) {
            int m = mi * 16 + hi * 8 + r;
            float v = acc[j][r];
            if (bias)  v += bias[col];
            if (mode == 1) v = gelu_exact(v);
            if (resid) v += resid[(size_t)m * N + col];
            if (Cf) Cf[(size_t)m * N + col] = v;
            if (Ch) Ch[(size_t)m * N + col] = (_Float16)v;
        }
    }
}

// ---------------------------------------------------------------- retention core, one block per (b,head)
__global__ void attn_kernel(const _Float16* __restrict__ qkvg, const float* __restrict__ gg,
                            const float* __restrict__ gb, float* __restrict__ Yc) {
    __shared__ float Ks[81 * 16], Vs[81 * 16];
    int bh = blockIdx.x;
    int head = bh & 3, b = bh >> 2;
    int tid = threadIdx.x;
    float q[16];
    if (tid < 81) {
        const _Float16* base = qkvg + (size_t)(b * 81 + tid) * 256 + head * 16;
        v8h q0 = *(const v8h*)(base),       q1 = *(const v8h*)(base + 8);
        v8h k0 = *(const v8h*)(base + 64),  k1 = *(const v8h*)(base + 72);
        v8h v0 = *(const v8h*)(base + 128), v1 = *(const v8h*)(base + 136);
        float pos = (float)tid;
#pragma unroll
        for (int j = 0; j < 8; ++j) {
            float scale = (2.f * j + 6.4f) / 22.4f;
            float xs = powf(scale, pos / 512.f);
            float ang = pos * powf(10000.f, -(float)j / 8.f);
            float c = cosf(ang), s = sinf(ang);
            float cu = c * xs, su = s * xs, cd = c / xs, sd = s / xs;
            float q1v = (j < 4) ? (float)q0[2*j]   : (float)q1[2*j-8];
            float q2v = (j < 4) ? (float)q0[2*j+1] : (float)q1[2*j-7];
            q[2*j]   = q1v * cu - q2v * su;
            q[2*j+1] = q2v * cu + q1v * su;
            float k1v = (j < 4) ? (float)k0[2*j]   : (float)k1[2*j-8];
            float k2v = (j < 4) ? (float)k0[2*j+1] : (float)k1[2*j-7];
            Ks[tid * 16 + 2*j]   = k1v * cd - k2v * sd;
            Ks[tid * 16 + 2*j+1] = k2v * cd + k1v * sd;
            Vs[tid * 16 + 2*j]   = (j < 4) ? (float)v0[2*j]   : (float)v1[2*j-8];
            Vs[tid * 16 + 2*j+1] = (j < 4) ? (float)v0[2*j+1] : (float)v1[2*j-7];
        }
    }
    __syncthreads();
    if (tid < 81) {
        float g = 1.f - expf(logf(1.f / 32.f) +
                             (float)head * (logf(1.f / 512.f) - logf(1.f / 32.f)) / 3.f);
        float y[16];
#pragma unroll
        for (int j = 0; j < 16; ++j) y[j] = 0.f;
        float w = 1.f;
        for (int t = tid; t >= 0; --t) {
            float dot = 0.f;
#pragma unroll
            for (int j = 0; j < 16; ++j) dot += q[j] * Ks[t * 16 + j];
            dot *= w;
#pragma unroll
            for (int j = 0; j < 16; ++j) y[j] += dot * Vs[t * 16 + j];
            w *= g;
        }
        float mu = 0.f;
#pragma unroll
        for (int j = 0; j < 16; ++j) mu += y[j];
        mu *= (1.f / 16.f);
        float var = 0.f;
#pragma unroll
        for (int j = 0; j < 16; ++j) { float d = y[j] - mu; var += d * d; }
        float rstd = rsqrtf(var * (1.f / 16.f) + EPS_);
#pragma unroll
        for (int j = 0; j < 16; ++j) {
            int cch = head * 16 + j;
            Yc[(size_t)(b * 81 + tid) * 64 + cch] = (y[j] - mu) * rstd * gg[cch] + gb[cch];
        }
    }
}

// silu(G) * Yc -> f16, 8-wide
__global__ void gate8(const _Float16* __restrict__ qkvg, const float* __restrict__ Yc,
                      _Float16* __restrict__ t) {
    int i = blockIdx.x * blockDim.x + threadIdx.x;
    if (i >= 82944 * 64 / 8) return;
    int base = i * 8;
    int c0 = base & 63; int r = base >> 6;
    v8h gH = *(const v8h*)(qkvg + (size_t)r * 256 + 192 + c0);
    float4 y0 = *(const float4*)(Yc + base);
    float4 y1 = *(const float4*)(Yc + base + 4);
    float yv[8] = {y0.x, y0.y, y0.z, y0.w, y1.x, y1.y, y1.z, y1.w};
    v8h o;
#pragma unroll
    for (int j = 0; j < 8; ++j) {
        float G = (float)gH[j];
        o[j] = (_Float16)((G / (1.f + expf(-G))) * yv[j]);
    }
    *(v8h*)(t + base) = o;
}

// ================================================================ host
static inline int gemm_blocks(int M, int N, int NT) {
    long long tiles = (long long)(M / 16) * ((N / 16) / NT);
    return (int)((tiles + 7) / 8);
}

extern "C" void kernel_launch(void* const* d_in, const int* in_sizes, int n_in,
                              void* d_out, int out_size, void* d_ws, size_t ws_size,
                              hipStream_t stream) {
    (void)in_sizes; (void)n_in; (void)out_size; (void)ws_size;
    const float* X    = (const float*)d_in[0];
    const float* c3w  = (const float*)d_in[1];
    const float* c3b  = (const float*)d_in[2];
    const float* bn3g = (const float*)d_in[3];
    const float* bn3b = (const float*)d_in[4];
    const float* c2w  = (const float*)d_in[5];
    const float* c2b  = (const float*)d_in[6];
    const float* bn2g = (const float*)d_in[7];
    const float* bn2b = (const float*)d_in[8];
    const float* wq   = (const float*)d_in[9];
    const float* wk   = (const float*)d_in[10];
    const float* wv   = (const float*)d_in[11];
    const float* wg   = (const float*)d_in[12];
    const float* wo   = (const float*)d_in[13];
    const float* gng  = (const float*)d_in[14];
    const float* gnb  = (const float*)d_in[15];
    const float* ln1g = (const float*)d_in[16];
    const float* ln1b = (const float*)d_in[17];
    const float* ln2g = (const float*)d_in[18];
    const float* ln2b = (const float*)d_in[19];
    const float* fw1  = (const float*)d_in[20];
    const float* fb1  = (const float*)d_in[21];
    const float* fw2  = (const float*)d_in[22];
    const float* fb2  = (const float*)d_in[23];
    const float* nn1w = (const float*)d_in[24];
    const float* nn1b = (const float*)d_in[25];

    char* wp = (char*)d_ws;
    auto alloc = [&](size_t bytes) -> void* {
        void* p = (void*)wp; wp += (bytes + 255) & ~(size_t)255; return p;
    };
    const size_t R = 82944;                                  // B*S
    float*     stats = (float*)alloc(1024);
    _Float16*  A2h   = (_Float16*)alloc((size_t)B_ * 121 * 224 * 2 + 256); // +slack for conv2 overshoot
    _Float16*  w2t   = (_Float16*)alloc((size_t)64 * 2304 * 2);
    float*     C2    = (float*)alloc(R * 64 * 4);
    float*     xbuf  = (float*)alloc(R * 64 * 4);
    _Float16*  hbuf  = (_Float16*)alloc(R * 64 * 2);
    _Float16*  qkvg  = (_Float16*)alloc(R * 256 * 2);
    float*     Ycb   = (float*)alloc(R * 64 * 4);
    _Float16*  tbuf  = (_Float16*)alloc(R * 64 * 2);
    float*     ybuf  = (float*)alloc(R * 64 * 4);
    _Float16*  f1h   = (_Float16*)alloc(R * 128 * 2);
    _Float16*  xh    = (_Float16*)alloc(R * 64 * 2);
    _Float16*  wcatT = (_Float16*)alloc((size_t)256 * 64 * 2);
    _Float16*  woT   = (_Float16*)alloc((size_t)64 * 64 * 2);
    _Float16*  fw1T  = (_Float16*)alloc((size_t)128 * 64 * 2);
    _Float16*  fw2T  = (_Float16*)alloc((size_t)64 * 128 * 2);
    _Float16*  nn1wT = (_Float16*)alloc((size_t)16 * 5184 * 2);

    const int c3tot = B_ * 8 * 28 * 121;
    conv3_kernel<<<(c3tot + 255) / 256, 256, 0, stream>>>(X, c3w, c3b, A2h);
    bn3_stats  <<<8, 256, 0, stream>>>(A2h, stats);
    bn3_apply  <<<(c3tot + 255) / 256, 256, 0, stream>>>(A2h, stats, bn3g, bn3b);
    pack_w2t   <<<(64 * 2304 + 255) / 256, 256, 0, stream>>>(c2w, w2t);
    conv2_wmma <<<(5184 + 7) / 8, 256, 0, stream>>>(A2h, w2t, c2b, C2);
    bn2_stats  <<<64, 256, 0, stream>>>(C2, stats);
    bn2_apply8 <<<(int)((R * 8 + 255) / 256), 256, 0, stream>>>(C2, stats, bn2g, bn2b, xbuf);

    for (int l = 0; l < 2; ++l) {
        ln_kernel<<<(int)((R + 255) / 256), 256, 0, stream>>>(xbuf, ln1g + l * 64, ln1b + l * 64, hbuf, (int)R);
        pack_wcat_t<<<(64 * 256 + 255) / 256, 256, 0, stream>>>(wq, wk, wv, wg, l, wcatT);
        gemm_wmma<4><<<gemm_blocks((int)R, 256, 4), 256, 0, stream>>>(hbuf, wcatT, nullptr, nullptr,
                                                                      nullptr, qkvg, (int)R, 256, 64, 0);
        attn_kernel<<<B_ * 4, 128, 0, stream>>>(qkvg, gng + l * 64, gnb + l * 64, Ycb);
        gate8<<<(int)((R * 8 + 255) / 256), 256, 0, stream>>>(qkvg, Ycb, tbuf);
        tr_cast<<<(4096 + 255) / 256, 256, 0, stream>>>(wo + l * 4096, woT, 64, 64);
        gemm_wmma<4><<<gemm_blocks((int)R, 64, 4), 256, 0, stream>>>(tbuf, woT, nullptr, xbuf,
                                                                     ybuf, nullptr, (int)R, 64, 64, 0);
        ln_kernel<<<(int)((R + 255) / 256), 256, 0, stream>>>(ybuf, ln2g + l * 64, ln2b + l * 64, hbuf, (int)R);
        tr_cast<<<(8192 + 255) / 256, 256, 0, stream>>>(fw1 + l * 8192, fw1T, 64, 128);
        gemm_wmma<4><<<gemm_blocks((int)R, 128, 4), 256, 0, stream>>>(hbuf, fw1T, fb1 + l * 128, nullptr,
                                                                      nullptr, f1h, (int)R, 128, 64, 1);
        tr_cast<<<(8192 + 255) / 256, 256, 0, stream>>>(fw2 + l * 8192, fw2T, 128, 64);
        gemm_wmma<4><<<gemm_blocks((int)R, 64, 4), 256, 0, stream>>>(f1h, fw2T, fb2 + l * 64, ybuf,
                                                                     xbuf, nullptr, (int)R, 64, 128, 0);
    }
    cast8<<<(int)((R * 8 + 255) / 256), 256, 0, stream>>>(xbuf, xh, (int)(R * 8));
    tr_cast<<<(5184 * 16 + 255) / 256, 256, 0, stream>>>(nn1w, nn1wT, 5184, 16);
    gemm_wmma<1><<<gemm_blocks(1024, 16, 1), 256, 0, stream>>>(xh, nn1wT, nn1b, nullptr,
                                                               (float*)d_out, nullptr, 1024, 16, 5184, 0);
}
